// CDBlock_47098611368284
// MI455X (gfx1250) — matrix operand order repaired and checked by reading, not compiled, not run on hardware
//
#include <hip/hip_runtime.h>
#include <math.h>

typedef __attribute__((ext_vector_type(16))) _Float16 v16h;
typedef __attribute__((ext_vector_type(8)))  float    v8f;

#define C_DIM  32
#define K_DIM  16
#define L_DIM  11
#define EPS_BN 1e-5f

__device__ __forceinline__ float leaky_f(float v, float s) { return v >= 0.f ? v : s * v; }

// ---------------------------------------------------------------------------
// Column-wise BatchNorm statistics -> affine (scale, shift) so that
// bn(x) = x*scale + shift.  One block per column.
// ---------------------------------------------------------------------------
__global__ __launch_bounds__(256) void col_stats_kernel(
    const float* __restrict__ X, int n_rows, int n_cols,
    const float* __restrict__ g, const float* __restrict__ b,
    float* __restrict__ scale, float* __restrict__ shift) {
  const int c = blockIdx.x;
  float s = 0.f, s2 = 0.f;
  for (int r = threadIdx.x; r < n_rows; r += blockDim.x) {
    float v = X[(size_t)r * n_cols + c];
    s += v;
    s2 += v * v;
  }
  __shared__ float sh[256], sh2[256];
  sh[threadIdx.x] = s;
  sh2[threadIdx.x] = s2;
  __syncthreads();
  for (int st = 128; st > 0; st >>= 1) {
    if (threadIdx.x < st) {
      sh[threadIdx.x] += sh[threadIdx.x + st];
      sh2[threadIdx.x] += sh2[threadIdx.x + st];
    }
    __syncthreads();
  }
  if (threadIdx.x == 0) {
    float inv_n = 1.f / (float)n_rows;
    float m = sh[0] * inv_n;
    float var = sh2[0] * inv_n - m * m;
    float istd = rsqrtf(var + EPS_BN);
    float sc = istd * g[c];
    scale[c] = sc;
    shift[c] = b[c] - m * sc;
  }
}

// ---------------------------------------------------------------------------
// WMMA fragment helpers (CDNA5 16x16x32 f16 layouts, wave32).
// A 16x32 (MxK): lanes 0-15 -> M=lane, K slots {k0..k0+7, k0+16..k0+23};
//                lanes 16-31 -> M=lane-16, K slots {k0+8..+15, k0+24..+31}.
// B 32x16 (KxN): lane -> N=col_base+(lane&15), K slots same as A for that lane half.
// C/D 16x16:     lane -> N=lane&15, VGPR r -> M=r+8*(lane>>4).
//
// Out-of-range A rows are CLAMPED (not zeroed): a bad A row only pollutes the
// matching D row, and D stores mask rows >= nrows.  This keeps every fragment
// load an unconditional b128.
// ---------------------------------------------------------------------------
template <bool BN>
__device__ __forceinline__ v16h load_a_tile(const float* __restrict__ X, int ldx,
                                            int row_base, int k0, int lane, int nrows,
                                            const float* __restrict__ scale,
                                            const float* __restrict__ shift,
                                            float slope) {
  int row = row_base + (lane & 15);
  row = row < nrows ? row : nrows - 1;
  const int koff = (lane & 16) ? 8 : 0;
  const float* rp = X + (size_t)row * ldx + (k0 + koff);
  float4 q0 = *(const float4*)(rp + 0);
  float4 q1 = *(const float4*)(rp + 4);
  float4 q2 = *(const float4*)(rp + 16);
  float4 q3 = *(const float4*)(rp + 20);
  float v[16] = {q0.x, q0.y, q0.z, q0.w, q1.x, q1.y, q1.z, q1.w,
                 q2.x, q2.y, q2.z, q2.w, q3.x, q3.y, q3.z, q3.w};
  if (BN) {
    const float* sp = scale + (k0 + koff);
    const float* fp = shift + (k0 + koff);
    float4 s0 = *(const float4*)(sp + 0), s1 = *(const float4*)(sp + 4);
    float4 s2 = *(const float4*)(sp + 16), s3 = *(const float4*)(sp + 20);
    float4 f0 = *(const float4*)(fp + 0), f1 = *(const float4*)(fp + 4);
    float4 f2 = *(const float4*)(fp + 16), f3 = *(const float4*)(fp + 20);
    float ss[16] = {s0.x, s0.y, s0.z, s0.w, s1.x, s1.y, s1.z, s1.w,
                    s2.x, s2.y, s2.z, s2.w, s3.x, s3.y, s3.z, s3.w};
    float ff[16] = {f0.x, f0.y, f0.z, f0.w, f1.x, f1.y, f1.z, f1.w,
                    f2.x, f2.y, f2.z, f2.w, f3.x, f3.y, f3.z, f3.w};
#pragma unroll
    for (int e = 0; e < 16; ++e) v[e] = leaky_f(v[e] * ss[e] + ff[e], slope);
  }
  v16h a;
#pragma unroll
  for (int e = 0; e < 16; ++e) a[e] = (_Float16)v[e];
  return a;
}

__device__ __forceinline__ v16h load_b_tile(const float* __restrict__ W, int ldw,
                                            int k0, int col_base, int lane) {
  const int koff = (lane & 16) ? 8 : 0;
  const float* rp = W + (size_t)(k0 + koff) * ldw + col_base + (lane & 15);
  float v[16];
#pragma unroll
  for (int e = 0; e < 8; ++e) {
    v[e] = rp[(size_t)e * ldw];
    v[e + 8] = rp[(size_t)(e + 16) * ldw];
  }
  v16h b;
#pragma unroll
  for (int e = 0; e < 16; ++e) b[e] = (_Float16)v[e];
  return b;
}

template <bool RESID>
__device__ __forceinline__ void store_c_tile(float* __restrict__ O, int ldo,
                                             int row_base, int col_base, int lane,
                                             int nrows, v8f c,
                                             const float* __restrict__ resid, int ldr) {
  const int n = lane & 15;
  const int mb = (lane & 16) ? 8 : 0;
  if (row_base + 16 <= nrows) {  // full stripe: unguarded
#pragma unroll
    for (int r = 0; r < 8; ++r) {
      int row = row_base + mb + r;
      float v = c[r];
      if (RESID) v += resid[(size_t)row * ldr + col_base + n];
      O[(size_t)row * ldo + col_base + n] = v;
    }
  } else {
#pragma unroll
    for (int r = 0; r < 8; ++r) {
      int row = row_base + mb + r;
      if (row < nrows) {
        float v = c[r];
        if (RESID) v += resid[(size_t)row * ldr + col_base + n];
        O[(size_t)row * ldo + col_base + n] = v;
      }
    }
  }
}

// ---------------------------------------------------------------------------
// GEMM 1: h_pre = leaky(bn1(x)) @ lin_in_W        [N x 128] @ [128 x 32]
// One wave per 16-row stripe; A fragment loaded once, used by both col tiles.
// ---------------------------------------------------------------------------
__global__ __launch_bounds__(256) void gemm_in_kernel(
    const float* __restrict__ X, const float* __restrict__ scale,
    const float* __restrict__ shift, const float* __restrict__ W,
    float* __restrict__ O, int nrows) {
  const int lane = threadIdx.x & 31;
  const int wave = threadIdx.x >> 5;
  const int nstripes = (nrows + 15) >> 4;
  const int stripe = blockIdx.x * 8 + wave;
  if (stripe >= nstripes) return;
  const int row_base = stripe << 4;
  v8f acc0 = {}, acc1 = {};
#pragma unroll
  for (int kb = 0; kb < 4; ++kb) {
    const int k0 = kb * 32;
    v16h a = load_a_tile<true>(X, 128, row_base, k0, lane, nrows, scale, shift, 0.1f);
    v16h b0 = load_b_tile(W, 32, k0, 0, lane);
    v16h b1 = load_b_tile(W, 32, k0, 16, lane);
    acc0 = __builtin_amdgcn_wmma_f32_16x16x32_f16(false, a, false, b0, (short)0, acc0,
                                                  false, false);
    acc1 = __builtin_amdgcn_wmma_f32_16x16x32_f16(false, a, false, b1, (short)0, acc1,
                                                  false, false);
  }
  store_c_tile<false>(O, 32, row_base, 0, lane, nrows, acc0, nullptr, 0);
  store_c_tile<false>(O, 32, row_base, 16, lane, nrows, acc1, nullptr, 0);
}

// ---------------------------------------------------------------------------
// Edge MLP: per-edge geometry + 2-layer per-seq-index MLP -> ws = w*smooth.
// Weights (17.6 KB) preloaded into LDS.
// ---------------------------------------------------------------------------
__global__ __launch_bounds__(256) void edge_mlp_kernel(
    const float* __restrict__ pos, const float* __restrict__ ori,
    const int* __restrict__ resno, const int* __restrict__ edges,
    const float* __restrict__ W0, const float* __restrict__ b0,
    const float* __restrict__ W1, const float* __restrict__ b1,
    float* __restrict__ ws_all, int E) {
  __shared__ float sW0[L_DIM * 7 * K_DIM];
  __shared__ float sb0[L_DIM * K_DIM];
  __shared__ float sW1[L_DIM * K_DIM * K_DIM];
  __shared__ float sb1[L_DIM * K_DIM];
  for (int i = threadIdx.x; i < L_DIM * 7 * K_DIM; i += blockDim.x) sW0[i] = W0[i];
  for (int i = threadIdx.x; i < L_DIM * K_DIM; i += blockDim.x) {
    sb0[i] = b0[i];
    sb1[i] = b1[i];
  }
  for (int i = threadIdx.x; i < L_DIM * K_DIM * K_DIM; i += blockDim.x) sW1[i] = W1[i];
  __syncthreads();

  const int e = blockIdx.x * blockDim.x + threadIdx.x;
  if (e >= E) return;
  const int ni = edges[2 * e];
  const int no = edges[2 * e + 1];

  float t0 = pos[3 * ni + 0] - pos[3 * no + 0];
  float t1 = pos[3 * ni + 1] - pos[3 * no + 1];
  float t2 = pos[3 * ni + 2] - pos[3 * no + 2];
  float dist = sqrtf(t0 * t0 + t1 * t1 + t2 * t2);
  float inv = 1.f / (dist + 1e-9f);
  t0 *= inv; t1 *= inv; t2 *= inv;

  const float* Oo = ori + (size_t)9 * no;
  const float* Oi = ori + (size_t)9 * ni;
  float delta[7];
  delta[0] = Oo[0] * t0 + Oo[1] * t1 + Oo[2] * t2;
  delta[1] = Oo[3] * t0 + Oo[4] * t1 + Oo[5] * t2;
  delta[2] = Oo[6] * t0 + Oo[7] * t1 + Oo[8] * t2;
  delta[3] = Oo[0] * Oi[0] + Oo[1] * Oi[1] + Oo[2] * Oi[2];
  delta[4] = Oo[3] * Oi[3] + Oo[4] * Oi[4] + Oo[5] * Oi[5];
  delta[5] = Oo[6] * Oi[6] + Oo[7] * Oi[7] + Oo[8] * Oi[8];
  delta[6] = dist;

  int sd = resno[ni] - resno[no];
  sd = sd < -5 ? -5 : (sd > 5 ? 5 : sd);
  const int sidx = sd + 5;
  const float nl = fabsf((float)sd) * 0.2f;

  const float* w0p = sW0 + sidx * 7 * K_DIM;
  const float* b0p = sb0 + sidx * K_DIM;
  float w[K_DIM];
#pragma unroll
  for (int o = 0; o < K_DIM; ++o) {
    float acc = b0p[o];
#pragma unroll
    for (int i = 0; i < 7; ++i) acc += delta[i] * w0p[i * K_DIM + o];
    w[o] = leaky_f(acc, 0.2f);
  }
  const float* w1p = sW1 + sidx * K_DIM * K_DIM;
  const float* b1p = sb1 + sidx * K_DIM;
  float w2[K_DIM];
#pragma unroll
  for (int o = 0; o < K_DIM; ++o) {
    float acc = b1p[o];
#pragma unroll
    for (int i = 0; i < K_DIM; ++i) acc += w[i] * w1p[i * K_DIM + o];
    w2[o] = leaky_f(acc, 0.2f);
  }

  const float nd = dist * 0.25f;  // dist / SPATIAL_CUTOFF
  const float smooth = 0.5f - 0.5f * tanhf(nd * nl * 16.f - 14.f);
  float4* op = (float4*)(ws_all + (size_t)e * K_DIM);
#pragma unroll
  for (int q = 0; q < 4; ++q) {
    float4 v = make_float4(w2[4 * q + 0] * smooth, w2[4 * q + 1] * smooth,
                           w2[4 * q + 2] * smooth, w2[4 * q + 3] * smooth);
    op[q] = v;
  }
}

// ---------------------------------------------------------------------------
// Scatter: agg[node_out, k*32+c] += ws[e,k] * leaky(bn2(h_pre[node_in,c])).
// 8 edges / block, lane = channel c -> 32 contiguous atomics per (edge,k),
// resolving in L2 (agg = 51 MB << 192 MB L2).  Relaxed agent-scope fadd ->
// non-returning global_atomic_add_f32.
// ---------------------------------------------------------------------------
__global__ __launch_bounds__(256) void edge_scatter_kernel(
    const int* __restrict__ edges, const float* __restrict__ ws_all,
    const float* __restrict__ h_pre, const float* __restrict__ scale2,
    const float* __restrict__ shift2, float* __restrict__ agg, int E) {
  const int lane = threadIdx.x & 31;
  const int sub = threadIdx.x >> 5;
  const int e = blockIdx.x * 8 + sub;
  if (e >= E) return;
  const int ni = edges[2 * e];
  const int no = edges[2 * e + 1];
  float h = h_pre[(size_t)ni * C_DIM + lane];
  h = leaky_f(h * scale2[lane] + shift2[lane], 0.1f);

  const float4* wsp = (const float4*)(ws_all + (size_t)e * K_DIM);
  float4 w0 = wsp[0], w1 = wsp[1], w2 = wsp[2], w3 = wsp[3];
  float wsk[K_DIM] = {w0.x, w0.y, w0.z, w0.w, w1.x, w1.y, w1.z, w1.w,
                      w2.x, w2.y, w2.z, w2.w, w3.x, w3.y, w3.z, w3.w};
  float* base = agg + (size_t)no * (K_DIM * C_DIM) + lane;
#pragma unroll
  for (int k = 0; k < K_DIM; ++k) {
    __hip_atomic_fetch_add(base + k * C_DIM, wsk[k] * h, __ATOMIC_RELAXED,
                           __HIP_MEMORY_SCOPE_AGENT);
  }
}

// ---------------------------------------------------------------------------
// GEMM 2: upd_pre = agg @ conv_W                  [N x 512] @ [512 x 32]
// One wave per stripe; agg (the dominant 51 MB stream) is read exactly once.
// ---------------------------------------------------------------------------
__global__ __launch_bounds__(256) void gemm_conv_kernel(
    const float* __restrict__ A, const float* __restrict__ W,
    float* __restrict__ O, int nrows) {
  const int lane = threadIdx.x & 31;
  const int wave = threadIdx.x >> 5;
  const int nstripes = (nrows + 15) >> 4;
  const int stripe = blockIdx.x * 8 + wave;
  if (stripe >= nstripes) return;
  const int row_base = stripe << 4;
  v8f acc0 = {}, acc1 = {};
#pragma unroll
  for (int kb = 0; kb < 16; ++kb) {
    const int k0 = kb * 32;
    v16h a = load_a_tile<false>(A, 512, row_base, k0, lane, nrows, nullptr, nullptr, 0.f);
    v16h b0 = load_b_tile(W, 32, k0, 0, lane);
    v16h b1 = load_b_tile(W, 32, k0, 16, lane);
    acc0 = __builtin_amdgcn_wmma_f32_16x16x32_f16(false, a, false, b0, (short)0, acc0,
                                                  false, false);
    acc1 = __builtin_amdgcn_wmma_f32_16x16x32_f16(false, a, false, b1, (short)0, acc1,
                                                  false, false);
  }
  store_c_tile<false>(O, 32, row_base, 0, lane, nrows, acc0, nullptr, 0);
  store_c_tile<false>(O, 32, row_base, 16, lane, nrows, acc1, nullptr, 0);
}

// ---------------------------------------------------------------------------
// GEMM 3: out = leaky(bn3(upd_pre)) @ lin_out_W + x   [N x 32] @ [32 x 128]
// One wave per stripe; A loaded once, 8 WMMAs (one per 16-col tile).
// ---------------------------------------------------------------------------
__global__ __launch_bounds__(256) void gemm_out_kernel(
    const float* __restrict__ U, const float* __restrict__ scale,
    const float* __restrict__ shift, const float* __restrict__ W,
    const float* __restrict__ X, float* __restrict__ O, int nrows) {
  const int lane = threadIdx.x & 31;
  const int wave = threadIdx.x >> 5;
  const int nstripes = (nrows + 15) >> 4;
  const int stripe = blockIdx.x * 8 + wave;
  if (stripe >= nstripes) return;
  const int row_base = stripe << 4;
  v16h a = load_a_tile<true>(U, 32, row_base, 0, lane, nrows, scale, shift, 0.1f);
#pragma unroll
  for (int ct = 0; ct < 8; ++ct) {
    const int col_base = ct * 16;
    v16h b = load_b_tile(W, 128, 0, col_base, lane);
    v8f acc = {};
    acc = __builtin_amdgcn_wmma_f32_16x16x32_f16(false, a, false, b, (short)0, acc,
                                                 false, false);
    store_c_tile<true>(O, 128, row_base, col_base, lane, nrows, acc, X, 128);
  }
}

// ---------------------------------------------------------------------------
// Host launcher
// ---------------------------------------------------------------------------
extern "C" void kernel_launch(void* const* d_in, const int* in_sizes, int n_in,
                              void* d_out, int out_size, void* d_ws, size_t ws_size,
                              hipStream_t stream) {
  const float* x        = (const float*)d_in[0];   // N x 128
  const float* pos      = (const float*)d_in[1];   // N x 3
  const float* ori      = (const float*)d_in[2];   // N x 3 x 3
  const int*   resno    = (const int*)  d_in[3];   // N
  const int*   edges    = (const int*)  d_in[4];   // E x 2
  const float* bn1_g    = (const float*)d_in[5];
  const float* bn1_b    = (const float*)d_in[6];
  const float* lin_in_W = (const float*)d_in[7];   // 128 x 32
  const float* bn2_g    = (const float*)d_in[8];
  const float* bn2_b    = (const float*)d_in[9];
  const float* wn_W0    = (const float*)d_in[10];  // 11 x 7 x 16
  const float* wn_b0    = (const float*)d_in[11];  // 11 x 16
  const float* wn_W1    = (const float*)d_in[12];  // 11 x 16 x 16
  const float* wn_b1    = (const float*)d_in[13];  // 11 x 16
  const float* conv_W   = (const float*)d_in[14];  // 512 x 32
  const float* bn3_g    = (const float*)d_in[15];
  const float* bn3_b    = (const float*)d_in[16];
  const float* lin_out_W= (const float*)d_in[17];  // 32 x 128
  float* out = (float*)d_out;

  const int N = in_sizes[0] / 128;
  const int E = in_sizes[4] / 2;

  // Workspace layout (all offsets 256B-aligned).
  char* ws = (char*)d_ws;
  size_t off = 0;
  auto alloc = [&](size_t bytes) {
    char* p = ws + off;
    off += (bytes + 255) & ~(size_t)255;
    return p;
  };
  float* scale1 = (float*)alloc(128 * sizeof(float));
  float* shift1 = (float*)alloc(128 * sizeof(float));
  float* scale2 = (float*)alloc(32 * sizeof(float));
  float* shift2 = (float*)alloc(32 * sizeof(float));
  float* scale3 = (float*)alloc(32 * sizeof(float));
  float* shift3 = (float*)alloc(32 * sizeof(float));
  float* h_pre  = (float*)alloc((size_t)N * 32 * sizeof(float));
  float* upd_pre= (float*)alloc((size_t)N * 32 * sizeof(float));
  float* ws_all = (float*)alloc((size_t)E * 16 * sizeof(float));
  float* agg    = (float*)alloc((size_t)N * 512 * sizeof(float));
  (void)ws_size;

  const int nstripes = (N + 15) >> 4;
  const int stripe_blocks = (nstripes + 7) / 8;

  // 1. BN1 stats (128 columns of x)
  col_stats_kernel<<<128, 256, 0, stream>>>(x, N, 128, bn1_g, bn1_b, scale1, shift1);

  // 2. h_pre = leaky(bn1(x)) @ lin_in_W   (WMMA)
  gemm_in_kernel<<<stripe_blocks, 256, 0, stream>>>(x, scale1, shift1, lin_in_W,
                                                    h_pre, N);

  // 3. BN2 stats (32 columns of h_pre)
  col_stats_kernel<<<32, 256, 0, stream>>>(h_pre, N, 32, bn2_g, bn2_b, scale2, shift2);

  // 4. zero agg (must happen every call; harness does not re-zero)
  hipMemsetAsync(agg, 0, (size_t)N * 512 * sizeof(float), stream);

  // 5. per-edge MLP -> ws_all
  edge_mlp_kernel<<<(E + 255) / 256, 256, 0, stream>>>(pos, ori, resno, edges, wn_W0,
                                                       wn_b0, wn_W1, wn_b1, ws_all, E);

  // 6. scatter outer products into agg (L2-resident atomics)
  edge_scatter_kernel<<<(E + 7) / 8, 256, 0, stream>>>(edges, ws_all, h_pre, scale2,
                                                       shift2, agg, E);

  // 7. upd_pre = agg @ conv_W   (WMMA, K=512, agg read once)
  gemm_conv_kernel<<<stripe_blocks, 256, 0, stream>>>(agg, conv_W, upd_pre, N);

  // 8. BN3 stats (32 columns of upd_pre)
  col_stats_kernel<<<32, 256, 0, stream>>>(upd_pre, N, 32, bn3_g, bn3_b, scale3, shift3);

  // 9. out = leaky(bn3(upd_pre)) @ lin_out_W + x   (WMMA)
  gemm_out_kernel<<<stripe_blocks, 256, 0, stream>>>(upd_pre, scale3, shift3,
                                                     lin_out_W, x, out, N);
}
